// CausalSelfAttention_1082331758972
// MI455X (gfx1250) — matrix operand order repaired and checked by reading, not compiled, not run on hardware
//
#include <hip/hip_runtime.h>
#include <hip/hip_bf16.h>

// Problem constants (match reference)
#define BB 4
#define TT 2048
#define CC 1024
#define HH 16
#define HD 64

typedef __attribute__((ext_vector_type(16))) __bf16 v16bf;
typedef __attribute__((ext_vector_type(8)))  float  v8f;

union UB16 {
  unsigned short u[16];
  uint4 q[2];
  v16bf v;
};

// fp32 -> bf16 via native cast (RNE); bitcast to ushort for storage
__device__ inline unsigned short f2bf(float f) {
  union { __bf16 b; unsigned short s; } u;
  u.b = (__bf16)f;
  return u.s;
}

// ---------------------------------------------------------------------------
// Kernel 0: streaming fp32 -> bf16 conversion (done ONCE so GEMM inner loops
// are pure load_b128 + wmma with no conversion VALU).
// ---------------------------------------------------------------------------
__global__ __launch_bounds__(256) void cvt_bf16(const float* __restrict__ in,
                                                unsigned short* __restrict__ out,
                                                int n4) {
  const int i = blockIdx.x * blockDim.x + threadIdx.x;
  if (i < n4) {
    const float4 f = ((const float4*)in)[i];
    union { unsigned short s[4]; uint2 v; } u;
    u.s[0] = f2bf(f.x); u.s[1] = f2bf(f.y);
    u.s[2] = f2bf(f.z); u.s[3] = f2bf(f.w);
    ((uint2*)out)[i] = u.v;
  }
}

// ---------------------------------------------------------------------------
// Kernel 1: qkv = x @ w_attn.T  (bf16 operands, f32 accum).
// One wave computes a 32x64 output block (2 M-tiles x 4 N-tiles = 8 wmma per
// 32-wide K step; A/B frags reused 4x/2x). blockIdx.y = 0/1/2 -> q/k/v so the
// store-path branch is scalar. Q,K stored [b][h][t][d]; V transposed
// [b][h][d][t] (all bf16).
// ---------------------------------------------------------------------------
__global__ __launch_bounds__(256) void qkv_gemm(const unsigned short* __restrict__ xb,
                                                const unsigned short* __restrict__ wab,
                                                unsigned short* __restrict__ qg,
                                                unsigned short* __restrict__ kg,
                                                unsigned short* __restrict__ vtg) {
  const int lane = threadIdx.x & 31;
  const int wave = blockIdx.x * 8 + (threadIdx.x >> 5);  // 0..4095
  const int part = blockIdx.y;                           // 0=q, 1=k, 2=v (scalar)
  const int mt = wave >> 4;                              // 0..255
  const int nb = wave & 15;                              // head index
  const int m0 = mt * 32;
  const int hf = lane >> 4, l16 = lane & 15;

  const unsigned short* xr0 = xb + (size_t)(m0 + l16) * CC;
  const unsigned short* xr1 = xb + (size_t)(m0 + 16 + l16) * CC;
  const unsigned short* wr  = wab + (size_t)(part * 1024 + nb * 64 + l16) * CC;

  v8f acc[2][4] = {};
  for (int k0 = 0; k0 < CC; k0 += 32) {
    UB16 a0, a1;
    // A 16x32 frag: element i -> K = hf*16 + i
    a0.q[0] = *(const uint4*)(xr0 + k0 + hf * 16);
    a0.q[1] = *(const uint4*)(xr0 + k0 + hf * 16 + 8);
    a1.q[0] = *(const uint4*)(xr1 + k0 + hf * 16);
    a1.q[1] = *(const uint4*)(xr1 + k0 + hf * 16 + 8);
#pragma unroll
    for (int ni = 0; ni < 4; ++ni) {
      UB16 b;  // B 32x16 frag: element i -> K = hf*16 + i, N = l16
      const unsigned short* w = wr + (size_t)ni * (16 * CC) + k0 + hf * 16;
      b.q[0] = *(const uint4*)(w);
      b.q[1] = *(const uint4*)(w + 8);
      acc[0][ni] = __builtin_amdgcn_wmma_f32_16x16x32_bf16(false, a0.v, false, b.v,
                                                           (short)0, acc[0][ni], false, false);
      acc[1][ni] = __builtin_amdgcn_wmma_f32_16x16x32_bf16(false, a1.v, false, b.v,
                                                           (short)0, acc[1][ni], false, false);
    }
  }

  // C frag mapping: VGPR r -> M = r + hf*8, N = l16
  if (part < 2) {
    unsigned short* dst = (part == 0) ? qg : kg;   // [b][h][t][d]
#pragma unroll
    for (int mi = 0; mi < 2; ++mi) {
#pragma unroll
      for (int r = 0; r < 8; ++r) {
        const int g = m0 + mi * 16 + r + hf * 8;
        const int bidx = g >> 11, t = g & 2047;
        unsigned short* p = dst + ((size_t)(bidx * HH + nb) * TT + t) * HD;
#pragma unroll
        for (int ni = 0; ni < 4; ++ni)
          p[ni * 16 + l16] = f2bf(acc[mi][ni][r]);
      }
    }
  } else {                                          // v -> [b][h][d][t]
#pragma unroll
    for (int mi = 0; mi < 2; ++mi) {
#pragma unroll
      for (int r = 0; r < 8; ++r) {
        const int g = m0 + mi * 16 + r + hf * 8;
        const int bidx = g >> 11, t = g & 2047;
#pragma unroll
        for (int ni = 0; ni < 4; ++ni)
          vtg[((size_t)(bidx * HH + nb) * HD + ni * 16 + l16) * TT + t] =
              f2bf(acc[mi][ni][r]);
      }
    }
  }
}

// ---------------------------------------------------------------------------
// Kernel 2: flash attention. One wave per (b, h, 16-row query tile).
// S = Q K^T (2 chained wmma per 16-key tile, K-dim = hd = 64), online softmax
// with __shfl_xor row reductions (width 16 = wave half), P transposed
// C-layout -> A-layout through per-wave LDS, then O += P V (K=32 per block).
// ---------------------------------------------------------------------------
__global__ __launch_bounds__(256) void attn(const unsigned short* __restrict__ qg,
                                            const unsigned short* __restrict__ kg,
                                            const unsigned short* __restrict__ vtg,
                                            unsigned short* __restrict__ yg) {
  __shared__ __align__(16) unsigned short plds[8][512];  // 16x32 bf16 per wave
  const int lane = threadIdx.x & 31;
  const int widx = threadIdx.x >> 5;
  const int wave = blockIdx.x * 8 + widx;
  const int qi = wave & 127;       // query tile
  const int bh = wave >> 7;        // b*16 + h
  const int t0 = qi * 16;
  const int hf = lane >> 4, l16 = lane & 15;

  const unsigned short* qb = qg  + (size_t)bh * (TT * HD);
  const unsigned short* kb = kg  + (size_t)bh * (TT * HD);
  const unsigned short* vb = vtg + (size_t)bh * (HD * TT);

  UB16 qa0, qa1;  // Q A-frags (16x64 = two 16x32 frags over d)
  {
    const unsigned short* qr = qb + (t0 + l16) * HD;
    qa0.q[0] = *(const uint4*)(qr + hf * 16);
    qa0.q[1] = *(const uint4*)(qr + hf * 16 + 8);
    qa1.q[0] = *(const uint4*)(qr + 32 + hf * 16);
    qa1.q[1] = *(const uint4*)(qr + 32 + hf * 16 + 8);
  }

  v8f o[4] = {};
  float mo[8], lsum[8];
#pragma unroll
  for (int r = 0; r < 8; ++r) { mo[r] = -3.0e38f; lsum[r] = 0.f; }

  unsigned short* pl = plds[widx];
  const int kbmax = (t0 + 15) >> 5;

  for (int kb2 = 0; kb2 <= kbmax; ++kb2) {
    const int j0 = kb2 * 32;

    v8f s0, s1;
    {
      UB16 b0, b1;
      const unsigned short* kr = kb + (j0 + l16) * HD;
      b0.q[0] = *(const uint4*)(kr + hf * 16);
      b0.q[1] = *(const uint4*)(kr + hf * 16 + 8);
      b1.q[0] = *(const uint4*)(kr + 32 + hf * 16);
      b1.q[1] = *(const uint4*)(kr + 32 + hf * 16 + 8);
      v8f z = {};
      z  = __builtin_amdgcn_wmma_f32_16x16x32_bf16(false, qa0.v, false, b0.v,
                                                   (short)0, z, false, false);
      s0 = __builtin_amdgcn_wmma_f32_16x16x32_bf16(false, qa1.v, false, b1.v,
                                                   (short)0, z, false, false);
    }
    {
      UB16 b0, b1;
      const unsigned short* kr = kb + (j0 + 16 + l16) * HD;
      b0.q[0] = *(const uint4*)(kr + hf * 16);
      b0.q[1] = *(const uint4*)(kr + hf * 16 + 8);
      b1.q[0] = *(const uint4*)(kr + 32 + hf * 16);
      b1.q[1] = *(const uint4*)(kr + 32 + hf * 16 + 8);
      v8f z = {};
      z  = __builtin_amdgcn_wmma_f32_16x16x32_bf16(false, qa0.v, false, b0.v,
                                                   (short)0, z, false, false);
      s1 = __builtin_amdgcn_wmma_f32_16x16x32_bf16(false, qa1.v, false, b1.v,
                                                   (short)0, z, false, false);
    }

    float p0[8], p1[8];
#pragma unroll
    for (int r = 0; r < 8; ++r) {
      const int rowq = t0 + r + hf * 8;
      const int c0 = j0 + l16, c1 = j0 + 16 + l16;
      const bool ok0 = (c0 <= rowq), ok1 = (c1 <= rowq);
      float v0 = ok0 ? s0[r] * 0.125f : -3.0e38f;
      float v1 = ok1 ? s1[r] * 0.125f : -3.0e38f;
      float mx = fmaxf(v0, v1);
      mx = fmaxf(mx, __shfl_xor(mx, 1, 16));
      mx = fmaxf(mx, __shfl_xor(mx, 2, 16));
      mx = fmaxf(mx, __shfl_xor(mx, 4, 16));
      mx = fmaxf(mx, __shfl_xor(mx, 8, 16));
      const float mn = fmaxf(mo[r], mx);
      const float e0 = ok0 ? __expf(v0 - mn) : 0.f;
      const float e1 = ok1 ? __expf(v1 - mn) : 0.f;
      float rs = e0 + e1;
      rs += __shfl_xor(rs, 1, 16);
      rs += __shfl_xor(rs, 2, 16);
      rs += __shfl_xor(rs, 4, 16);
      rs += __shfl_xor(rs, 8, 16);
      const float corr = __expf(mo[r] - mn);
      lsum[r] = lsum[r] * corr + rs;
      mo[r] = mn;
      o[0][r] *= corr; o[1][r] *= corr; o[2][r] *= corr; o[3][r] *= corr;
      p0[r] = e0; p1[r] = e1;
    }

    // transpose P (C-layout) -> A-frag via per-wave LDS tile
#pragma unroll
    for (int r = 0; r < 8; ++r) {
      const int rq = r + hf * 8;
      pl[rq * 32 + l16]      = f2bf(p0[r]);
      pl[rq * 32 + 16 + l16] = f2bf(p1[r]);
    }
    asm volatile("s_wait_dscnt 0" ::: "memory");
    UB16 pf;
    pf.q[0] = *(const uint4*)(pl + l16 * 32 + hf * 16);
    pf.q[1] = *(const uint4*)(pl + l16 * 32 + hf * 16 + 8);

#pragma unroll
    for (int dt = 0; dt < 4; ++dt) {
      UB16 vf;  // B frag from V^T: contiguous keys per lane
      const unsigned short* vr = vb + (size_t)(dt * 16 + l16) * TT + j0 + hf * 16;
      vf.q[0] = *(const uint4*)(vr);
      vf.q[1] = *(const uint4*)(vr + 8);
      o[dt] = __builtin_amdgcn_wmma_f32_16x16x32_bf16(false, pf.v, false, vf.v,
                                                      (short)0, o[dt], false, false);
    }
  }

  const int bidx = bh >> 4, h = bh & 15;
#pragma unroll
  for (int r = 0; r < 8; ++r) {
    const int t = t0 + r + hf * 8;
    const float inv = 1.0f / lsum[r];
    unsigned short* yr = yg + (size_t)(bidx * TT + t) * CC + h * HD;
#pragma unroll
    for (int dt = 0; dt < 4; ++dt)
      yr[dt * 16 + l16] = f2bf(o[dt][r] * inv);
  }
}

// ---------------------------------------------------------------------------
// Kernel 3: out = y @ w_proj.T (all-bf16 operands, fp32 output).
// Same 32x64-per-wave tiling as kernel 1.
// ---------------------------------------------------------------------------
__global__ __launch_bounds__(256) void proj_gemm(const unsigned short* __restrict__ yg,
                                                 const unsigned short* __restrict__ wpb,
                                                 float* __restrict__ out) {
  const int lane = threadIdx.x & 31;
  const int wave = blockIdx.x * 8 + (threadIdx.x >> 5);  // 0..4095
  const int mt = wave >> 4;                              // 0..255
  const int nb = wave & 15;                              // 64-col block
  const int m0 = mt * 32, n0 = nb * 64;
  const int hf = lane >> 4, l16 = lane & 15;

  const unsigned short* yr0 = yg + (size_t)(m0 + l16) * CC;
  const unsigned short* yr1 = yg + (size_t)(m0 + 16 + l16) * CC;
  const unsigned short* wr  = wpb + (size_t)(n0 + l16) * CC;

  v8f acc[2][4] = {};
  for (int k0 = 0; k0 < CC; k0 += 32) {
    UB16 a0, a1;
    a0.q[0] = *(const uint4*)(yr0 + k0 + hf * 16);
    a0.q[1] = *(const uint4*)(yr0 + k0 + hf * 16 + 8);
    a1.q[0] = *(const uint4*)(yr1 + k0 + hf * 16);
    a1.q[1] = *(const uint4*)(yr1 + k0 + hf * 16 + 8);
#pragma unroll
    for (int ni = 0; ni < 4; ++ni) {
      UB16 b;
      const unsigned short* w = wr + (size_t)ni * (16 * CC) + k0 + hf * 16;
      b.q[0] = *(const uint4*)(w);
      b.q[1] = *(const uint4*)(w + 8);
      acc[0][ni] = __builtin_amdgcn_wmma_f32_16x16x32_bf16(false, a0.v, false, b.v,
                                                           (short)0, acc[0][ni], false, false);
      acc[1][ni] = __builtin_amdgcn_wmma_f32_16x16x32_bf16(false, a1.v, false, b.v,
                                                           (short)0, acc[1][ni], false, false);
    }
  }
#pragma unroll
  for (int mi = 0; mi < 2; ++mi) {
#pragma unroll
    for (int r = 0; r < 8; ++r) {
      const size_t g = m0 + mi * 16 + r + hf * 8;
#pragma unroll
      for (int ni = 0; ni < 4; ++ni)
        out[g * CC + n0 + ni * 16 + l16] = acc[mi][ni][r];
    }
  }
}

// ---------------------------------------------------------------------------
extern "C" void kernel_launch(void* const* d_in, const int* in_sizes, int n_in,
                              void* d_out, int out_size, void* d_ws, size_t ws_size,
                              hipStream_t stream) {
  const float* x   = (const float*)d_in[0];   // [4,2048,1024]
  const float* wat = (const float*)d_in[1];   // [3072,1024]
  const float* wp  = (const float*)d_in[2];   // [1024,1024]
  float* out = (float*)d_out;                 // [4,2048,1024]

  const size_t NELE = (size_t)BB * HH * TT * HD;      // 8,388,608
  unsigned short* q   = (unsigned short*)d_ws;
  unsigned short* k   = q + NELE;
  unsigned short* vt  = k + NELE;                     // V^T [b][h][d][t]
  unsigned short* y   = vt + NELE;                    // attn out bf16 [b][t][c]
  unsigned short* xbb = y + NELE;                     // x bf16
  unsigned short* wab = xbb + NELE;                   // w_attn bf16 (3072*1024)
  unsigned short* wpb = wab + (size_t)3072 * 1024;    // w_proj bf16 (1024*1024)

  // one-time fp32 -> bf16 conversions (streaming, trivially bandwidth-bound)
  cvt_bf16<<<8192, 256, 0, stream>>>(x,   xbb, (int)(NELE / 4));
  cvt_bf16<<<3072, 256, 0, stream>>>(wat, wab, 3072 * 1024 / 4);
  cvt_bf16<<<1024, 256, 0, stream>>>(wp,  wpb, 1024 * 1024 / 4);

  // 3 parts x 4096 waves (256 M-blocks x 16 heads), 8 waves/block
  qkv_gemm<<<dim3(512, 3), 256, 0, stream>>>(xbb, wab, q, k, vt);
  // 64 (b,h) x 128 query tiles = 8192 waves
  attn<<<1024, 256, 0, stream>>>(q, k, vt, y);
  // 4096 waves
  proj_gemm<<<512, 256, 0, stream>>>(y, wpb, out);
}
